// LlamaAttentionExperimental_30863634989508
// MI455X (gfx1250) — matrix operand order repaired and checked by reading, not compiled, not run on hardware
//
#include <hip/hip_runtime.h>
#include <hip/hip_bf16.h>

// ---------------------------------------------------------------------------
// LlamaAttention (B=2,S=2048,HID=4096, NH=32, NKV=8, D=128, GQA x4, RoPE,
// causal) for MI455X / gfx1250, wave32.
//   - all GEMM FLOPs through V_WMMA_F32_16X16X32_BF16 (f32 accumulate)
//   - bf16-resident tiles staged LDS via GLOBAL_LOAD_ASYNC_TO_LDS_B128
//   - f32->bf16 converted tiles staged via VALU + global_prefetch_b8
// ---------------------------------------------------------------------------

#define S_LEN 2048
#define HID_  4096
#define D_HEAD 128
#define NH_   32
#define NKV_  8

typedef __attribute__((ext_vector_type(16))) __bf16 v16bf;
typedef __attribute__((ext_vector_type(8)))  float  v8f;

union Frag {              // one WMMA A/B operand: 8 VGPRs = 16 bf16 per lane
  v16bf v;
  unsigned u[8];
};

__device__ __forceinline__ unsigned short f2bfu(float f) {
  union { float f; unsigned u; } x; x.f = f;
  unsigned u = x.u;
  u += 0x7fffu + ((u >> 16) & 1u);      // round-to-nearest-even
  return (unsigned short)(u >> 16);
}
__device__ __forceinline__ unsigned pack2bf(float lo, float hi) {
  return (unsigned)f2bfu(lo) | ((unsigned)f2bfu(hi) << 16);
}

// Async global->LDS copy, 16 bytes per lane, tracked by ASYNCcnt.
// GVS mode: mem_addr = SADDR64 + VADDR32 + inst_offset. LDS dest comes from
// the low 32 bits of the flat shared pointer (flat->LDS maps addr[31:0]).
__device__ __forceinline__ void async_copy_b128(void* lds_ptr, const void* gbase,
                                                int goff) {
  unsigned lds_off = (unsigned)(size_t)lds_ptr;
  asm volatile("global_load_async_to_lds_b128 %0, %1, %2"
               :: "v"(lds_off), "v"(goff), "s"(gbase)
               : "memory");
}
__device__ __forceinline__ void wait_async0() {
  asm volatile("s_wait_asynccnt 0x0" ::: "memory");
}

// ---------------------------------------------------------------------------
// Kernel 1: Out(M x N) = X(f32, M x K) * W(f32, N x K)^T ; optional fused RoPE
// epilogue; bf16 output in (B, H, S, D) layout.
// Tile: 128(M) x 128(N), 8 waves, each wave 32x64 -> 8 WMMA per K-step.
// ---------------------------------------------------------------------------
__global__ __launch_bounds__(256)
void gemm_qkv_rope(const float* __restrict__ X, const float* __restrict__ W,
                   const float* __restrict__ cosT, const float* __restrict__ sinT,
                   unsigned short* __restrict__ outp, int N, int applyRope)
{
  __shared__ unsigned lds_x[128][17];   // [row][K-pair]  (bf16 pairs)
  __shared__ unsigned lds_w[128][17];   // [n  ][K-pair]
  __shared__ float    lds_c[128][128];  // epilogue exchange for RoPE

  const int tid  = threadIdx.x;
  const int wave = tid >> 5;
  const int lane = tid & 31;
  const int l15  = lane & 15;
  const int half = lane >> 4;

  const int m_base = blockIdx.y * 128;
  const int n_base = blockIdx.x * 128;
  const int m0 = (wave & 3) * 32;       // wave's 32-row strip (2 A-frags)
  const int n0 = (wave >> 2) * 64;      // wave's 64-col strip (4 N-tiles)

  v8f acc[2][4];
  #pragma unroll
  for (int i = 0; i < 2; ++i)
    #pragma unroll
    for (int j = 0; j < 4; ++j) acc[i][j] = {};

  for (int k0 = 0; k0 < HID_; k0 += 32) {
    __syncthreads();
    // stage X tile (128 x 32 f32 -> bf16 pairs): 2048 pairs / 256 thr
    #pragma unroll
    for (int j = 0; j < 8; ++j) {
      int p = tid + j * 256;
      int row = p >> 4, c = p & 15;
      const float* src = X + (size_t)(m_base + row) * HID_ + k0 + 2 * c;
      lds_x[row][c] = pack2bf(src[0], src[1]);
      if (k0 + 32 < HID_) __builtin_prefetch(src + 32, 0, 1);
    }
    // stage W tile (128 x 32): 2048 pairs / 256 thr
    #pragma unroll
    for (int j = 0; j < 8; ++j) {
      int p = tid + j * 256;
      int row = p >> 4, c = p & 15;
      const float* src = W + (size_t)(n_base + row) * HID_ + k0 + 2 * c;
      lds_w[row][c] = pack2bf(src[0], src[1]);
      if (k0 + 32 < HID_) __builtin_prefetch(src + 32, 0, 1);
    }
    __syncthreads();

    // A fragments: rows m0+mi*16..+15, K interleave {0-7,16-23}/{8-15,24-31}
    Frag a[2];
    #pragma unroll
    for (int mi = 0; mi < 2; ++mi) {
      int r = m0 + mi * 16 + l15;
      #pragma unroll
      for (int v = 0; v < 8; ++v) {
        int p = (v < 4 ? v : v + 4) + (half ? 4 : 0);
        a[mi].u[v] = lds_x[r][p];
      }
    }
    #pragma unroll
    for (int nt = 0; nt < 4; ++nt) {
      Frag b;
      int n = n0 + nt * 16 + l15;
      #pragma unroll
      for (int v = 0; v < 8; ++v) b.u[v] = lds_w[n][(v < 4 ? v : v + 4) + (half ? 4 : 0)];
      #pragma unroll
      for (int mi = 0; mi < 2; ++mi)
        acc[mi][nt] = __builtin_amdgcn_wmma_f32_16x16x32_bf16(
            false, a[mi].v, false, b.v, (short)0, acc[mi][nt], false, false);
    }
  }

  __syncthreads();
  // dump C to LDS so the RoPE epilogue can pair column d with d^64
  #pragma unroll
  for (int mi = 0; mi < 2; ++mi)
    #pragma unroll
    for (int nt = 0; nt < 4; ++nt)
      #pragma unroll
      for (int r = 0; r < 8; ++r)
        lds_c[m0 + mi * 16 + r + (half ? 8 : 0)][n0 + nt * 16 + l15] = acc[mi][nt][r];
  __syncthreads();

  const int H = N >> 7;   // heads in this projection
  for (int j = 0; j < 64; ++j) {
    int p   = tid + j * 256;
    int row = p >> 7;          // 0..127
    int col = p & 127;         // 0..127 (= head dim d)
    int m = m_base + row;
    int b = m >> 11, s = m & (S_LEN - 1);
    int n = n_base + col;
    int h = n >> 7, d = n & 127;
    float val = lds_c[row][col];
    float o;
    if (applyRope) {
      float partner = lds_c[row][col ^ 64];
      size_t cs = ((size_t)(b * S_LEN + s)) * D_HEAD + d;
      float cv = cosT[cs], sv = sinT[cs];
      float rot = (d < 64) ? -partner : partner;
      o = val * cv + rot * sv;
    } else {
      o = val;
    }
    outp[(((size_t)b * H + h) * S_LEN + s) * D_HEAD + d] = f2bfu(o);
  }
}

// ---------------------------------------------------------------------------
// Kernel 2: flash attention. Block = (qb, h, b), 4 waves x 16 q-rows,
// 64-key blocks, online softmax, bf16 WMMA for QK^T and PV.
// K block staged with GLOBAL_LOAD_ASYNC_TO_LDS_B128.
// ---------------------------------------------------------------------------
__global__ __launch_bounds__(128)
void attn_kernel(const unsigned short* __restrict__ qbf,
                 const unsigned short* __restrict__ kbf,
                 const unsigned short* __restrict__ vbf,
                 unsigned short* __restrict__ obf)
{
  __shared__ unsigned lds_k [64][68];        // [key][d-pair]; 272B row, 16B aligned
  __shared__ unsigned lds_vt[128][33];       // [d][key-pair]  (V transposed)
  __shared__ unsigned short lds_p[4][16][66];// per-wave P bounce (C->A layout)

  const int tid  = threadIdx.x;
  const int wave = tid >> 5;
  const int lane = tid & 31;
  const int l15  = lane & 15;
  const int half = lane >> 4;

  const int qb = blockIdx.x;
  const int h  = blockIdx.y;
  const int b  = blockIdx.z;
  const int hkv = h >> 2;                    // GROUPS = 4

  const int q0 = qb * 64 + wave * 16;

  const unsigned* qbase = (const unsigned*)qbf + (size_t)(b * NH_  + h  ) * S_LEN * 64;
  const char*     kbyte = (const char*)kbf + (size_t)(b * NKV_ + hkv) * S_LEN * 256;
  const unsigned short* vbase = vbf + (size_t)(b * NKV_ + hkv) * S_LEN * 128;

  // Q fragments for this wave's 16 rows (4 d-chunks of 32)
  Frag qf[4];
  {
    const unsigned* qp = qbase + (size_t)(q0 + l15) * 64;
    #pragma unroll
    for (int dc = 0; dc < 4; ++dc)
      #pragma unroll
      for (int v = 0; v < 8; ++v) {
        int p = dc * 16 + (v < 4 ? v : v + 4) + (half ? 4 : 0);
        qf[dc].u[v] = qp[p];
      }
  }

  v8f oacc[8];
  #pragma unroll
  for (int i = 0; i < 8; ++i) oacc[i] = {};
  float mst[8], lst[8];
  #pragma unroll
  for (int r = 0; r < 8; ++r) { mst[r] = -1e30f; lst[r] = 0.f; }

  const float scale = 0.08838834764831845f;  // 1/sqrt(128)

  for (int kb = 0; kb <= qb; ++kb) {
    const int k0 = kb * 64;
    __syncthreads();
    // stage K block (64 keys x 256B) asynchronously: 1024 x B128, 8 per thread
    #pragma unroll
    for (int j = 0; j < 8; ++j) {
      int p = tid + j * 128;
      int row = p >> 4, seg = p & 15;
      async_copy_b128(&lds_k[row][seg * 4], kbyte,
                      (k0 + row) * 256 + seg * 16);
    }
    // stage V^T (128 d x 32 key-pairs) -- transpose, element-wise
    for (int j = 0; j < 32; ++j) {
      int p = tid + j * 128;
      int d = p >> 5, kk = p & 31;
      unsigned lo = vbase[(size_t)(k0 + 2 * kk    ) * 128 + d];
      unsigned hi = vbase[(size_t)(k0 + 2 * kk + 1) * 128 + d];
      lds_vt[d][kk] = lo | (hi << 16);
    }
    wait_async0();
    __syncthreads();

    // ---- scores: S = Q * K^T  (16 WMMA) ----
    v8f sacc[4];
    #pragma unroll
    for (int i = 0; i < 4; ++i) sacc[i] = {};
    #pragma unroll
    for (int dc = 0; dc < 4; ++dc)
      #pragma unroll
      for (int nt = 0; nt < 4; ++nt) {
        Frag bk;
        int n = nt * 16 + l15;
        #pragma unroll
        for (int v = 0; v < 8; ++v) bk.u[v] = lds_k[n][dc * 16 + (v < 4 ? v : v + 4) + (half ? 4 : 0)];
        sacc[nt] = __builtin_amdgcn_wmma_f32_16x16x32_bf16(
            false, qf[dc].v, false, bk.v, (short)0, sacc[nt], false, false);
      }

    // ---- scale + causal mask (diagonal block only) ----
    const bool diag = (kb == qb);
    #pragma unroll
    for (int nt = 0; nt < 4; ++nt)
      #pragma unroll
      for (int r = 0; r < 8; ++r) {
        float sv = sacc[nt][r] * scale;
        if (diag) {
          int qrow = q0 + r + (half ? 8 : 0);
          int key  = k0 + nt * 16 + l15;
          if (key > qrow) sv = -1e30f;
        }
        sacc[nt][r] = sv;
      }

    // ---- online softmax: per-row max / sum over 16-lane halves ----
    #pragma unroll
    for (int r = 0; r < 8; ++r) {
      float m = fmaxf(fmaxf(sacc[0][r], sacc[1][r]), fmaxf(sacc[2][r], sacc[3][r]));
      #pragma unroll
      for (int off = 1; off < 16; off <<= 1) m = fmaxf(m, __shfl_xor(m, off, 32));
      float mn = fmaxf(mst[r], m);
      float sc_old = __expf(mst[r] - mn);
      mst[r] = mn;
      float sum = 0.f;
      #pragma unroll
      for (int nt = 0; nt < 4; ++nt) {
        float e = __expf(sacc[nt][r] - mn);
        sacc[nt][r] = e;
        sum += e;
      }
      #pragma unroll
      for (int off = 1; off < 16; off <<= 1) sum += __shfl_xor(sum, off, 32);
      lst[r] = lst[r] * sc_old + sum;
      #pragma unroll
      for (int dt = 0; dt < 8; ++dt) oacc[dt][r] *= sc_old;
    }

    // ---- P: C-layout -> A-layout via per-wave LDS bounce (bf16) ----
    #pragma unroll
    for (int nt = 0; nt < 4; ++nt)
      #pragma unroll
      for (int r = 0; r < 8; ++r)
        lds_p[wave][r + (half ? 8 : 0)][nt * 16 + l15] = f2bfu(sacc[nt][r]);

    // ---- O += P * V  (16 WMMA) ----
    #pragma unroll
    for (int kc = 0; kc < 2; ++kc) {
      Frag a;
      #pragma unroll
      for (int v = 0; v < 8; ++v) {
        int p = kc * 16 + (v < 4 ? v : v + 4) + (half ? 4 : 0);
        a.u[v] = *(const unsigned*)&lds_p[wave][l15][2 * p];
      }
      #pragma unroll
      for (int dt = 0; dt < 8; ++dt) {
        Frag bv;
        int d = dt * 16 + l15;
        #pragma unroll
        for (int v = 0; v < 8; ++v) bv.u[v] = lds_vt[d][kc * 16 + (v < 4 ? v : v + 4) + (half ? 4 : 0)];
        oacc[dt] = __builtin_amdgcn_wmma_f32_16x16x32_bf16(
            false, a.v, false, bv.v, (short)0, oacc[dt], false, false);
      }
    }
  }

  // ---- epilogue: O /= l, write bf16 attn output (B, S, NH*D) ----
  #pragma unroll
  for (int dt = 0; dt < 8; ++dt)
    #pragma unroll
    for (int r = 0; r < 8; ++r) {
      int q = q0 + r + (half ? 8 : 0);
      int d = dt * 16 + l15;
      float val = oacc[dt][r] / lst[r];
      obf[(size_t)(b * S_LEN + q) * (NH_ * D_HEAD) + h * D_HEAD + d] = f2bfu(val);
    }
}

// ---------------------------------------------------------------------------
// Kernel 3: Out(f32, M x N) = Attn(bf16, M x K) * Wo(f32, N x K)^T
// Tile 128x128, 8 waves x (32x64). A tile async-staged (already bf16).
// ---------------------------------------------------------------------------
__global__ __launch_bounds__(256)
void gemm_out(const unsigned short* __restrict__ Abf,
              const float* __restrict__ W, float* __restrict__ outp)
{
  __shared__ unsigned lds_x[128][20];   // 80B row stride: 16B aligned, no conflicts
  __shared__ unsigned lds_w[128][17];

  const int tid  = threadIdx.x;
  const int wave = tid >> 5;
  const int lane = tid & 31;
  const int l15  = lane & 15;
  const int half = lane >> 4;

  const int m_base = blockIdx.y * 128;
  const int n_base = blockIdx.x * 128;
  const int m0 = (wave & 3) * 32;
  const int n0 = (wave >> 2) * 64;

  const char* Abyte = (const char*)Abf + (size_t)m_base * HID_ * 2;

  v8f acc[2][4];
  #pragma unroll
  for (int i = 0; i < 2; ++i)
    #pragma unroll
    for (int j = 0; j < 4; ++j) acc[i][j] = {};

  for (int k0 = 0; k0 < HID_; k0 += 32) {
    __syncthreads();
    // stage A tile (128 rows x 64B) asynchronously: 512 x B128, 2 per thread
    #pragma unroll
    for (int j = 0; j < 2; ++j) {
      int p = tid + j * 256;
      int row = p >> 2, seg = p & 3;
      async_copy_b128(&lds_x[row][seg * 4], Abyte,
                      row * (HID_ * 2) + k0 * 2 + seg * 16);
    }
    // stage W tile (f32 -> bf16): 2048 pairs
    #pragma unroll
    for (int j = 0; j < 8; ++j) {
      int p = tid + j * 256;
      int row = p >> 4, c = p & 15;
      const float* src = W + (size_t)(n_base + row) * HID_ + k0 + 2 * c;
      lds_w[row][c] = pack2bf(src[0], src[1]);
      if (k0 + 32 < HID_) __builtin_prefetch(src + 32, 0, 1);
    }
    wait_async0();
    __syncthreads();

    Frag a[2];
    #pragma unroll
    for (int mi = 0; mi < 2; ++mi) {
      int r = m0 + mi * 16 + l15;
      #pragma unroll
      for (int v = 0; v < 8; ++v) {
        int p = (v < 4 ? v : v + 4) + (half ? 4 : 0);
        a[mi].u[v] = lds_x[r][p];
      }
    }
    #pragma unroll
    for (int nt = 0; nt < 4; ++nt) {
      Frag b;
      int n = n0 + nt * 16 + l15;
      #pragma unroll
      for (int v = 0; v < 8; ++v) b.u[v] = lds_w[n][(v < 4 ? v : v + 4) + (half ? 4 : 0)];
      #pragma unroll
      for (int mi = 0; mi < 2; ++mi)
        acc[mi][nt] = __builtin_amdgcn_wmma_f32_16x16x32_bf16(
            false, a[mi].v, false, b.v, (short)0, acc[mi][nt], false, false);
    }
  }

  #pragma unroll
  for (int mi = 0; mi < 2; ++mi)
    #pragma unroll
    for (int nt = 0; nt < 4; ++nt)
      #pragma unroll
      for (int r = 0; r < 8; ++r) {
        int row = m_base + m0 + mi * 16 + r + (half ? 8 : 0);
        int col = n_base + n0 + nt * 16 + l15;
        outp[(size_t)row * HID_ + col] = acc[mi][nt][r];
      }
}

// ---------------------------------------------------------------------------
extern "C" void kernel_launch(void* const* d_in, const int* in_sizes, int n_in,
                              void* d_out, int out_size, void* d_ws, size_t ws_size,
                              hipStream_t stream) {
  (void)in_sizes; (void)n_in; (void)out_size; (void)ws_size;
  const float* X    = (const float*)d_in[0];   // hidden_states (B,S,HID)
  const float* cosT = (const float*)d_in[1];   // (B,S,D)
  const float* sinT = (const float*)d_in[2];
  const float* Wq   = (const float*)d_in[3];   // (NH*D, HID)
  const float* Wk   = (const float*)d_in[4];   // (NKV*D, HID)
  const float* Wv   = (const float*)d_in[5];
  const float* Wo   = (const float*)d_in[6];   // (HID, NH*D)
  float* out = (float*)d_out;

  char* ws = (char*)d_ws;
  unsigned short* qbf = (unsigned short*)ws;                                  // 32 MB
  unsigned short* kbf = (unsigned short*)(ws + (size_t)32 * 1024 * 1024);     //  8 MB
  unsigned short* vbf = (unsigned short*)(ws + (size_t)40 * 1024 * 1024);     //  8 MB
  unsigned short* abf = (unsigned short*)(ws + (size_t)48 * 1024 * 1024);     // 32 MB

  dim3 blk(256);
  gemm_qkv_rope<<<dim3(32, 32), blk, 0, stream>>>(X, Wq, cosT, sinT, qbf, 4096, 1);
  gemm_qkv_rope<<<dim3(8,  32), blk, 0, stream>>>(X, Wk, cosT, sinT, kbf, 1024, 1);
  gemm_qkv_rope<<<dim3(8,  32), blk, 0, stream>>>(X, Wv, cosT, sinT, vbf, 1024, 0);
  attn_kernel<<<dim3(32, 32, 2), dim3(128), 0, stream>>>(qbf, kbf, vbf, abf);
  gemm_out<<<dim3(32, 32), blk, 0, stream>>>(abf, Wo, out);
}
